// M5_EncoderDecoder_v2_75788992905892
// MI455X (gfx1250) — matrix operand order repaired and checked by reading, not compiled, not run on hardware
//
#include <hip/hip_runtime.h>
#include <hip/hip_bf16.h>
#include <cstdint>
#include <cstddef>

#define B_ 256
#define T_ 512
#define H_ 512
#define STEPS_ 28

typedef __attribute__((ext_vector_type(16))) __bf16 v16bf;
typedef __attribute__((ext_vector_type(8)))  float  v8f;
typedef unsigned short u16;
typedef unsigned int   u32;

__device__ __forceinline__ float bf2f(u16 x) { return __uint_as_float(((u32)x) << 16); }
__device__ __forceinline__ u16 f2bf(float f) {
    u32 u = __float_as_uint(f);
    u32 r = (u + 0x7FFFu + ((u >> 16) & 1u)) >> 16;   // round-to-nearest-even
    return (u16)r;
}
__device__ __forceinline__ float sigm(float x) { return 1.f / (1.f + expf(-x)); }

union FragU { v16bf v; u32 u[8]; };

// A fragment: 16x32 bf16 tile, rows striped across lanes.
// lane<16: M=lane, K = k0+{0..15}; lane>=16: M=lane-16, K = k0+{16..31}
__device__ __forceinline__ v16bf load_a(const u16* A, int lda, int row, int k0, int lane) {
    int r  = row + (lane & 15);
    int kk = k0 + ((lane >> 4) << 4);
    const u32* p0 = (const u32*)(A + (size_t)r * lda + kk);
    const u32* p1 = (const u32*)(A + (size_t)r * lda + kk + 8);
    FragU f;
#pragma unroll
    for (int i = 0; i < 4; ++i) { f.u[i] = p0[i]; f.u[4 + i] = p1[i]; }
    return f.v;
}
// B fragment: 32x16 bf16 (KxN), element (k,n) = W[n,k] with W row-major [N,K].
// lane<16: N=lane, K=k0..k0+15 contiguous; lane>=16: N=lane-16, K=k0+16..k0+31
__device__ __forceinline__ v16bf load_b(const u16* W, int ldw, int col, int k0, int lane) {
    int c  = col + (lane & 15);
    int kk = k0 + ((lane >> 4) << 4);
    const u32* p = (const u32*)(W + (size_t)c * ldw + kk);
    FragU f;
#pragma unroll
    for (int i = 0; i < 8; ++i) f.u[i] = p[i];
    return f.v;
}
__device__ __forceinline__ v8f wmma_bf16(v16bf a, v16bf b, v8f c) {
    return __builtin_amdgcn_wmma_f32_16x16x32_bf16(false, a, false, b, (short)0, c, false, false);
}

// ---------------- elementwise helpers ----------------
__global__ void k_f32_to_bf16(const float* __restrict__ s, u16* __restrict__ d, size_t n) {
    size_t i = ((size_t)blockIdx.x * blockDim.x + threadIdx.x) * 4;
#pragma unroll
    for (int j = 0; j < 4; ++j) { size_t k = i + j; if (k < n) d[k] = f2bf(s[k]); }
}
__global__ void k_copy_acc_bf16(const u16* __restrict__ s, u16* __restrict__ d, int n, int init) {
    int i = blockIdx.x * blockDim.x + threadIdx.x;
    if (i >= n) return;
    float v = bf2f(s[i]);
    if (!init) v += bf2f(d[i]);
    d[i] = f2bf(v);
}
__global__ void k_zero_f32(float* p, int n) {
    int i = blockIdx.x * blockDim.x + threadIdx.x;
    if (i < n) p[i] = 0.f;
}

// ---------------- generic WMMA GEMM: out = A[M,K] @ W[N,K]^T + bias ----------------
// 2x2 register blocking: each wave owns a 32x32 output block (4 WMMA tiles).
// Requires M%32==0, N%32==0, K%32==0.
__global__ __launch_bounds__(256) void k_gemm_bf16(
        const u16* __restrict__ A, const u16* __restrict__ W,
        const float* __restrict__ bias, void* __restrict__ out,
        int M, int N, int K, int relu, int out_bf) {
    int w   = blockIdx.x * (blockDim.x >> 5) + (threadIdx.x >> 5);
    int nT2 = N >> 5;
    int tiles = (M >> 5) * nT2;
    if (w >= tiles) return;
    int tr = w / nT2, tc = w % nT2;
    int rBase = tr << 5, cBase = tc << 5;
    int lane = threadIdx.x & 31;
    v8f a00 = {}, a01 = {}, a10 = {}, a11 = {};
    for (int k0 = 0; k0 < K; k0 += 32) {
        v16bf fa0 = load_a(A, K, rBase,      k0, lane);
        v16bf fa1 = load_a(A, K, rBase + 16, k0, lane);
        v16bf fb0 = load_b(W, K, cBase,      k0, lane);
        v16bf fb1 = load_b(W, K, cBase + 16, k0, lane);
        a00 = wmma_bf16(fa0, fb0, a00);
        a01 = wmma_bf16(fa0, fb1, a01);
        a10 = wmma_bf16(fa1, fb0, a10);
        a11 = wmma_bf16(fa1, fb1, a11);
    }
    int col = cBase + (lane & 15);
    int rb  = rBase + ((lane >> 4) << 3);
    float bs0 = bias ? bias[col] : 0.f;
    float bs1 = bias ? bias[col + 16] : 0.f;
#pragma unroll
    for (int v = 0; v < 8; ++v) {
        float x00 = a00[v] + bs0;
        float x01 = a01[v] + bs1;
        float x10 = a10[v] + bs0;
        float x11 = a11[v] + bs1;
        if (relu) {
            x00 = fmaxf(x00, 0.f); x01 = fmaxf(x01, 0.f);
            x10 = fmaxf(x10, 0.f); x11 = fmaxf(x11, 0.f);
        }
        size_t i0 = (size_t)(rb + v) * N + col;
        size_t i1 = (size_t)(rb + 16 + v) * N + col;
        if (out_bf) {
            ((u16*)out)[i0]      = f2bf(x00);
            ((u16*)out)[i0 + 16] = f2bf(x01);
            ((u16*)out)[i1]      = f2bf(x10);
            ((u16*)out)[i1 + 16] = f2bf(x11);
        } else {
            ((float*)out)[i0]      = x00;
            ((float*)out)[i0 + 16] = x01;
            ((float*)out)[i1]      = x10;
            ((float*)out)[i1 + 16] = x11;
        }
    }
}

// ---------------- fused GRU recurrent step: gh = h@whh^T, gates + h update ----------------
// gi already holds x@wih^T + bih (bf16, [B,3H]). PyTorch order (r,z,n).
__global__ __launch_bounds__(256) void k_gru_gates(
        const u16* __restrict__ gi, const u16* __restrict__ hprev,
        const u16* __restrict__ whh, const float* __restrict__ bhh,
        u16* __restrict__ hnew, u16* __restrict__ outp,
        int outStride, int outOffset, int accum) {
    int w = blockIdx.x * (blockDim.x >> 5) + (threadIdx.x >> 5); // 0..511
    int tr = w >> 5, tc = w & 31;
    int rBase = tr << 4, cBase = tc << 4;
    int lane = threadIdx.x & 31;
    v8f ar = {}, az = {}, an = {};
    for (int k0 = 0; k0 < H_; k0 += 32) {
        v16bf a  = load_a(hprev, H_, rBase, k0, lane);
        v16bf br = load_b(whh, H_, cBase,         k0, lane);
        v16bf bz = load_b(whh, H_, cBase + H_,    k0, lane);
        v16bf bn = load_b(whh, H_, cBase + 2*H_,  k0, lane);
        ar = wmma_bf16(a, br, ar);
        az = wmma_bf16(a, bz, az);
        an = wmma_bf16(a, bn, an);
    }
    int j  = cBase + (lane & 15);
    int rb = rBase + ((lane >> 4) << 3);
    float br_ = bhh[j], bz_ = bhh[j + H_], bn_ = bhh[j + 2 * H_];
#pragma unroll
    for (int v = 0; v < 8; ++v) {
        int b = rb + v;
        size_t gidx = (size_t)b * (3 * H_) + j;
        float gir = bf2f(gi[gidx]);
        float giz = bf2f(gi[gidx + H_]);
        float gin = bf2f(gi[gidx + 2 * H_]);
        float r = sigm(gir + ar[v] + br_);
        float z = sigm(giz + az[v] + bz_);
        float n = tanhf(gin + r * (an[v] + bn_));
        float hp = bf2f(hprev[(size_t)b * H_ + j]);
        float hv = (1.f - z) * n + z * hp;
        hnew[(size_t)b * H_ + j] = f2bf(hv);
        if (outp) {
            size_t oidx = (size_t)b * outStride + outOffset + j;
            float o = hv;
            if (accum) o += bf2f(outp[oidx]);
            outp[oidx] = f2bf(o);
        }
    }
}

// ---------------- attention kernels ----------------
// e[b*T+t] = V . tanh(Upre[t,b,:] + Wh[b,:]) + Vb ; one wave per (t,b)
__global__ void k_attn_energy(const u16* __restrict__ Upre, const u16* __restrict__ Wh,
                              const float* __restrict__ Vw, const float* __restrict__ Vb,
                              float* __restrict__ e, int T, int B, int H) {
    int w = blockIdx.x * (blockDim.x >> 5) + (threadIdx.x >> 5);
    if (w >= T * B) return;
    int t = w / B, b = w % B;
    int lane = threadIdx.x & 31;
    const u16* up = Upre + ((size_t)t * B + b) * H;
    const u16* wh = Wh + (size_t)b * H;
    float s = 0.f;
    for (int h = lane; h < H; h += 32)
        s += tanhf(bf2f(up[h]) + bf2f(wh[h])) * Vw[h];
    for (int off = 16; off > 0; off >>= 1) s += __shfl_down(s, off, 32);
    if (lane == 0) e[(size_t)b * T + t] = s + Vb[0];
}
__global__ void k_softmax_T(const float* __restrict__ e, float* __restrict__ alpha, int T) {
    __shared__ float sm[T_];
    int b = blockIdx.x, t = threadIdx.x;
    float v = e[(size_t)b * T + t];
    sm[t] = v; __syncthreads();
    for (int s = T_ / 2; s > 0; s >>= 1) { if (t < s) sm[t] = fmaxf(sm[t], sm[t + s]); __syncthreads(); }
    float mx = sm[0]; __syncthreads();
    float ex = expf(v - mx);
    sm[t] = ex; __syncthreads();
    for (int s = T_ / 2; s > 0; s >>= 1) { if (t < s) sm[t] += sm[t + s]; __syncthreads(); }
    alpha[(size_t)b * T + t] = ex / sm[0];
}
// attn[b,h] = sum_t alpha[t]*enc_out[t,b,h] -> din[b, 64+h]
__global__ void k_attn_ctx(const float* __restrict__ alpha, const u16* __restrict__ enc_out,
                           u16* __restrict__ din, int T, int B, int H) {
    __shared__ float al[T_];
    int b = blockIdx.x, h = threadIdx.x;
    al[h] = alpha[(size_t)b * T + h]; __syncthreads();
    float acc = 0.f;
    for (int t = 0; t < T; ++t)
        acc += al[t] * bf2f(enc_out[((size_t)t * B + b) * H + h]);
    din[(size_t)b * 576 + 64 + h] = f2bf(acc);
}
__global__ void k_din_head(const float* __restrict__ fb, const float* __restrict__ xt,
                           u16* __restrict__ din) {
    int b = blockIdx.x, i = threadIdx.x; // 64 threads
    float v = (i == 0) ? fb[b] : xt[(size_t)b * 63 + (i - 1)];
    din[(size_t)b * 576 + i] = f2bf(v);
}
// out[b,o] = relu(hn[b,:] . h2o_w[o,:] + b[o]); fb[b] = out[b,0]; one wave per (b,o)
__global__ void k_h2o(const u16* __restrict__ hn, const float* __restrict__ W,
                      const float* __restrict__ bias, float* __restrict__ out,
                      float* __restrict__ fb, int B, int H) {
    int w = blockIdx.x * (blockDim.x >> 5) + (threadIdx.x >> 5);
    if (w >= B * 2) return;
    int b = w >> 1, o = w & 1;
    int lane = threadIdx.x & 31;
    float s = 0.f;
    for (int h = lane; h < H; h += 32)
        s += bf2f(hn[(size_t)b * H + h]) * W[(size_t)o * H + h];
    for (int off = 16; off > 0; off >>= 1) s += __shfl_down(s, off, 32);
    if (lane == 0) {
        float r = fmaxf(s + bias[o], 0.f);
        out[(size_t)b * 2 + o] = r;
        if (o == 0) fb[b] = r;
    }
}

// =======================================================================
extern "C" void kernel_launch(void* const* d_in, const int* in_sizes, int n_in,
                              void* d_out, int out_size, void* d_ws, size_t ws_size,
                              hipStream_t stream) {
    (void)in_sizes; (void)n_in; (void)out_size; (void)ws_size;
    const float* ann_w1 = (const float*)d_in[0];
    const float* ann_b1 = (const float*)d_in[1];
    const float* ann_w2 = (const float*)d_in[2];
    const float* ann_b2 = (const float*)d_in[3];
    const float* dec_wih = (const float*)d_in[4];
    const float* dec_whh = (const float*)d_in[5];
    const float* dec_bih = (const float*)d_in[6];
    const float* dec_bhh = (const float*)d_in[7];
    const float* U_w = (const float*)d_in[8];
    const float* U_b = (const float*)d_in[9];
    const float* W_w = (const float*)d_in[10];
    const float* W_b = (const float*)d_in[11];
    const float* V_w = (const float*)d_in[12];
    const float* V_b = (const float*)d_in[13];
    const float* h2o_w = (const float*)d_in[14];
    const float* h2o_b = (const float*)d_in[15];
    const float* wihP[4] = {(const float*)d_in[16], (const float*)d_in[20], (const float*)d_in[24], (const float*)d_in[28]};
    const float* whhP[4] = {(const float*)d_in[17], (const float*)d_in[21], (const float*)d_in[25], (const float*)d_in[29]};
    const float* bihP[4] = {(const float*)d_in[18], (const float*)d_in[22], (const float*)d_in[26], (const float*)d_in[30]};
    const float* bhhP[4] = {(const float*)d_in[19], (const float*)d_in[23], (const float*)d_in[27], (const float*)d_in[31]};
    const float* ann_data = (const float*)d_in[32];
    const float* enc_data = (const float*)d_in[33];
    const float* dec_data = (const float*)d_in[34];
    float* out = (float*)d_out;

    const size_t TB = (size_t)T_ * B_;
    char* wsBase = (char*)d_ws;
    size_t off = 0;
    auto alloc = [&](size_t bytes) -> char* {
        char* p = wsBase + off;
        off += (bytes + 255) & ~(size_t)255;
        return p;
    };

    u16* enc_x = (u16*)alloc(TB * 64 * 2);        // encoder input bf16
    u16* x1    = (u16*)alloc(TB * 1024 * 2);      // [T,B,2H] layer1 input
    u16* enc_o = (u16*)alloc(TB * H_ * 2);        // o1f + o1b
    u16* gi    = (u16*)alloc(TB * 3 * H_ * 2);    // precomputed input gates (reused per scan)
    u16* Upre  = (u16*)alloc(TB * H_ * 2);
    int wihK[4] = {64, 64, 1024, 1024};
    u16* w_wih[4]; for (int i = 0; i < 4; ++i) w_wih[i] = (u16*)alloc((size_t)3 * H_ * wihK[i] * 2);
    u16* w_whh[4]; for (int i = 0; i < 4; ++i) w_whh[i] = (u16*)alloc((size_t)3 * H_ * H_ * 2);
    u16* w_U    = (u16*)alloc((size_t)H_ * H_ * 2);
    u16* w_W    = (u16*)alloc((size_t)H_ * H_ * 2);
    u16* w_dwih = (u16*)alloc((size_t)3 * H_ * 576 * 2);
    u16* w_dwhh = (u16*)alloc((size_t)3 * H_ * H_ * 2);
    u16* w_a1   = (u16*)alloc(96 * 64 * 2);
    u16* w_a2   = (u16*)alloc((size_t)H_ * 96 * 2);
    u16* annA   = (u16*)alloc((size_t)B_ * 64 * 2);
    u16* t1     = (u16*)alloc((size_t)B_ * 96 * 2);
    u16* h0     = (u16*)alloc((size_t)B_ * H_ * 2);
    u16* hb0    = (u16*)alloc((size_t)B_ * H_ * 2);
    u16* hb1    = (u16*)alloc((size_t)B_ * H_ * 2);
    u16* hidden = (u16*)alloc((size_t)B_ * H_ * 2);
    u16* hd0    = (u16*)alloc((size_t)B_ * H_ * 2);
    u16* hd1    = (u16*)alloc((size_t)B_ * H_ * 2);
    u16* Wh     = (u16*)alloc((size_t)B_ * H_ * 2);
    float* e     = (float*)alloc((size_t)B_ * T_ * 4);
    float* alpha = (float*)alloc((size_t)B_ * T_ * 4);
    u16* din    = (u16*)alloc((size_t)B_ * 576 * 2);
    u16* gi_d   = (u16*)alloc((size_t)B_ * 3 * H_ * 2);
    float* fb   = (float*)alloc(B_ * 4);

    auto cv = [&](const float* s, u16* d, size_t n) {
        int blocks = (int)((n + 1023) / 1024);
        k_f32_to_bf16<<<blocks, 256, 0, stream>>>(s, d, n);
    };
    cv(ann_w1, w_a1, 96 * 64);
    cv(ann_w2, w_a2, (size_t)H_ * 96);
    for (int i = 0; i < 4; ++i) {
        cv(wihP[i], w_wih[i], (size_t)3 * H_ * wihK[i]);
        cv(whhP[i], w_whh[i], (size_t)3 * H_ * H_);
    }
    cv(U_w, w_U, (size_t)H_ * H_);
    cv(W_w, w_W, (size_t)H_ * H_);
    cv(dec_wih, w_dwih, (size_t)3 * H_ * 576);
    cv(dec_whh, w_dwhh, (size_t)3 * H_ * H_);
    cv(enc_data, enc_x, TB * 64);
    cv(ann_data, annA, (size_t)B_ * 64);

    auto gemm = [&](const u16* A, const u16* Wt, const float* bias, void* o,
                    int M, int N, int K, int relu, int obf) {
        int tiles = (M >> 5) * (N >> 5);          // 32x32 blocks, one wave each
        int blocks = (tiles + 7) >> 3;
        k_gemm_bf16<<<blocks, 256, 0, stream>>>(A, Wt, bias, o, M, N, K, relu, obf);
    };

    // ANN -> h0 (shared initial hidden)
    gemm(annA, w_a1, ann_b1, t1, B_, 96, 64, 1, 1);
    gemm(t1, w_a2, ann_b2, h0, B_, H_, 96, 0, 1);

    auto scan = [&](int li, u16* outBase, size_t outTStr, int outStride, int outOffset,
                    int accum, bool rev, int first) {
        k_copy_acc_bf16<<<(B_ * H_ + 255) / 256, 256, 0, stream>>>(h0, hb0, B_ * H_, 1);
        for (int s = 0; s < T_; ++s) {
            int t = rev ? (T_ - 1 - s) : s;
            const u16* hp = (s & 1) ? hb1 : hb0;
            u16* hn = (s & 1) ? hb0 : hb1;
            k_gru_gates<<<64, 256, 0, stream>>>(gi + (size_t)t * B_ * 3 * H_, hp,
                                                w_whh[li], bhhP[li], hn,
                                                outBase + (size_t)t * outTStr,
                                                outStride, outOffset, accum);
        }
        // 512 steps (even) -> final hidden lands in hb0
        k_copy_acc_bf16<<<(B_ * H_ + 255) / 256, 256, 0, stream>>>(hb0, hidden, B_ * H_, first);
    };

    // Encoder: layer0 fwd/bwd -> x1 ; layer1 fwd/bwd -> enc_o (summed)
    gemm(enc_x, w_wih[0], bihP[0], gi, (int)TB, 3 * H_, 64, 0, 1);
    scan(0, x1, (size_t)B_ * 1024, 1024, 0, 0, false, 1);
    gemm(enc_x, w_wih[1], bihP[1], gi, (int)TB, 3 * H_, 64, 0, 1);
    scan(1, x1, (size_t)B_ * 1024, 1024, H_, 0, true, 0);
    gemm(x1, w_wih[2], bihP[2], gi, (int)TB, 3 * H_, 1024, 0, 1);
    scan(2, enc_o, (size_t)B_ * H_, H_, 0, 0, false, 0);
    gemm(x1, w_wih[3], bihP[3], gi, (int)TB, 3 * H_, 1024, 0, 1);
    scan(3, enc_o, (size_t)B_ * H_, H_, 0, 1, true, 0);

    // Upre = enc_out @ U^T + U_b
    gemm(enc_o, w_U, U_b, Upre, (int)TB, H_, H_, 0, 1);

    // Decoder
    k_copy_acc_bf16<<<(B_ * H_ + 255) / 256, 256, 0, stream>>>(hidden, hd0, B_ * H_, 1);
    k_zero_f32<<<1, 256, 0, stream>>>(fb, B_);

    for (int s = 0; s < STEPS_; ++s) {
        const u16* hp = (s & 1) ? hd1 : hd0;
        u16* hn = (s & 1) ? hd0 : hd1;
        gemm(hp, w_W, W_b, Wh, B_, H_, H_, 0, 1);
        k_attn_energy<<<(T_ * B_ + 7) / 8, 256, 0, stream>>>(Upre, Wh, V_w, V_b, e, T_, B_, H_);
        k_softmax_T<<<B_, T_, 0, stream>>>(e, alpha, T_);
        k_attn_ctx<<<B_, H_, 0, stream>>>(alpha, enc_o, din, T_, B_, H_);
        k_din_head<<<B_, 64, 0, stream>>>(fb, dec_data + (size_t)s * B_ * 63, din);
        gemm(din, w_dwih, dec_bih, gi_d, B_, 3 * H_, 576, 0, 1);
        k_gru_gates<<<64, 256, 0, stream>>>(gi_d, hp, w_dwhh, dec_bhh, hn,
                                            (u16*)nullptr, 0, 0, 0);
        k_h2o<<<64, 256, 0, stream>>>(hn, h2o_w, h2o_b, out + (size_t)s * B_ * 2, fb, B_, H_);
    }
}
